// GNN_13494787244371
// MI455X (gfx1250) — matrix operand order repaired and checked by reading, not compiled, not run on hardware
//
#include <hip/hip_runtime.h>
#include <hip/hip_bf16.h>

typedef __attribute__((ext_vector_type(16))) _Float16 v16h;
typedef __attribute__((ext_vector_type(8)))  float    v8f;

#define N_NODES 100000
#define N_EDGES 3200000
#define N_GRAPHS 128
#define IN_DIM 4
#define HID 64
#define BN_EPS 1e-5f
// node rows padded to a multiple of 128 so the WMMA epilogue is branchless
#define NPAD (((N_NODES + 127) / 128) * 128)

// ---------------------------------------------------------------------------
// Degree
// ---------------------------------------------------------------------------
__global__ void degree_count_kernel(const int* __restrict__ dst, float* __restrict__ deg, int e) {
    int t = blockIdx.x * blockDim.x + threadIdx.x;
    if (t < e) atomicAdd(&deg[dst[t]], 1.0f);
}

__global__ void deg_inv_kernel(float* __restrict__ deg, int n) {
    int t = blockIdx.x * blockDim.x + threadIdx.x;
    if (t < n) deg[t] = 1.0f / fmaxf(deg[t], 1.0f);
}

// ---------------------------------------------------------------------------
// Edge scatter: msum[dst] += h[src]   (4-dim for layer 1, 64-dim for 2/3)
// ---------------------------------------------------------------------------
__global__ void scatter4_kernel(const float* __restrict__ x,
                                const int* __restrict__ src, const int* __restrict__ dst,
                                float* __restrict__ msum, int e) {
    int t = blockIdx.x * blockDim.x + threadIdx.x;
    if (t >= e) return;
    int s = src[t], d = dst[t];
    float4 v = *(const float4*)(x + (size_t)s * 4);
    float* o = msum + (size_t)d * 4;
    atomicAdd(o + 0, v.x); atomicAdd(o + 1, v.y);
    atomicAdd(o + 2, v.z); atomicAdd(o + 3, v.w);
}

__global__ void scatter64_kernel(const float* __restrict__ h,
                                 const int* __restrict__ src, const int* __restrict__ dst,
                                 float* __restrict__ msum, int e) {
    int t = blockIdx.x * blockDim.x + threadIdx.x;   // 16 threads per edge
    int eid  = t >> 4;
    if (eid >= e) return;
    int part = t & 15;
    int s = src[eid], d = dst[eid];
    float4 v = *(const float4*)(h + (size_t)s * HID + part * 4);
    float* o = msum + (size_t)d * HID + part * 4;
    atomicAdd(o + 0, v.x); atomicAdd(o + 1, v.y);
    atomicAdd(o + 2, v.z); atomicAdd(o + 3, v.w);
}

// ---------------------------------------------------------------------------
// Layer 1 (IN_DIM=4): scalar GEMM, z = (msum*dinv)@w1l.T + b1 + x@w1r.T
// ---------------------------------------------------------------------------
__global__ void sage1_kernel(const float* __restrict__ x, const float* __restrict__ msum4,
                             const float* __restrict__ dinv,
                             const float* __restrict__ w1l, const float* __restrict__ b1,
                             const float* __restrict__ w1r,
                             float* __restrict__ z, int n) {
    int t = blockIdx.x * blockDim.x + threadIdx.x;
    if (t >= n * HID) return;
    int nid = t >> 6, o = t & 63;
    float s = dinv[nid];
    float4 m  = *(const float4*)(msum4 + (size_t)nid * 4);
    float4 xr = *(const float4*)(x + (size_t)nid * 4);
    float4 wl = *(const float4*)(w1l + o * 4);
    float4 wr = *(const float4*)(w1r + o * 4);
    float acc = b1[o];
    acc += s * (m.x * wl.x + m.y * wl.y + m.z * wl.z + m.w * wl.w);
    acc += xr.x * wr.x + xr.y * wr.y + xr.z * wr.z + xr.w * wr.w;
    z[t] = acc;
}

// ---------------------------------------------------------------------------
// Pack [wl | wr] (each [64][64] f32) into f16 combined W [64][128]
// ---------------------------------------------------------------------------
__global__ void pack_w_kernel(const float* __restrict__ wl, const float* __restrict__ wr,
                              _Float16* __restrict__ Wc) {
    int t = blockIdx.x * blockDim.x + threadIdx.x;
    if (t >= HID * 128) return;
    int o = t >> 7, k = t & 127;
    float v = (k < HID) ? wl[o * HID + k] : wr[o * HID + (k - HID)];
    Wc[t] = (_Float16)v;
}

// ---------------------------------------------------------------------------
// Fused SAGE GEMM (layers 2/3): z = [agg | h] @ Wc.T + bias, via WMMA f16->f32
//   A = [N x 128] f16 built on the fly (agg = msum*dinv in K[0,64), h in K[64,128))
//   Wc staged in LDS; B fragments via ds_load_b128. D tiles routed through an
//   LDS transpose buffer -> branchless coalesced global_store_b128 write-back.
//   Block = 256 thr = 8 waves; wave -> 16 rows x 64 cols = 4 tiles x 4 WMMAs.
// ---------------------------------------------------------------------------
#define ZSTRIDE 68   // floats per LDS row: 16B-aligned, bank-conflict free

__global__ void __launch_bounds__(256)
sage_gemm_wmma_kernel(const float* __restrict__ msum, const float* __restrict__ dinv,
                      const float* __restrict__ h, const _Float16* __restrict__ Wc,
                      const float* __restrict__ bias, float* __restrict__ z, int n) {
    __shared__ _Float16 lW[HID * 128];            // 16 KB staged weights
    __shared__ float    lZ[8 * 16 * ZSTRIDE];     // per-wave 16x64 D-tile buffers

    const int tid  = threadIdx.x;
    const int lane = tid & 31;
    const int wave = tid >> 5;
    const int row0 = blockIdx.x * 128 + wave * 16;
    const int hi   = lane >> 4;        // half-wave select
    const int rloc = lane & 15;        // A row (M), B column (N) within tile

    // Cooperative weight staging: 4096 dwords, 4 x b128 per thread
    {
        const uint4* wsrc = (const uint4*)Wc;
        uint4*       wdst = (uint4*)lW;
#pragma unroll
        for (int i = 0; i < 4; ++i)
            wdst[tid + 256 * i] = wsrc[tid + 256 * i];
    }
    __builtin_prefetch(bias, 0, 1);
    __syncthreads();

    int rowm = row0 + rloc;
    if (rowm >= n) rowm = n - 1;       // clamp A reads for OOB rows
    const float s = dinv[rowm];
    const float* mrow = msum + (size_t)rowm * HID;
    const float* hrow = h    + (size_t)rowm * HID;

    // A fragments: lane holds row M=rloc; elem j -> K = kb+16*hi+j
    //              (contiguous order, matches B fragments)
    v16h a[4];
#pragma unroll
    for (int f = 0; f < 4; ++f) {
        const float* srcp = (f < 2) ? mrow : hrow;
        const float  sc   = (f < 2) ? s : 1.0f;
        const int kb = (f & 1) * 32;
        const float4 c0  = *(const float4*)(srcp + kb + 16 * hi);
        const float4 c0b = *(const float4*)(srcp + kb + 16 * hi + 4);
        const float4 c1  = *(const float4*)(srcp + kb + 16 * hi + 8);
        const float4 c1b = *(const float4*)(srcp + kb + 16 * hi + 12);
        a[f][0]  = (_Float16)(c0.x * sc);  a[f][1]  = (_Float16)(c0.y * sc);
        a[f][2]  = (_Float16)(c0.z * sc);  a[f][3]  = (_Float16)(c0.w * sc);
        a[f][4]  = (_Float16)(c0b.x * sc); a[f][5]  = (_Float16)(c0b.y * sc);
        a[f][6]  = (_Float16)(c0b.z * sc); a[f][7]  = (_Float16)(c0b.w * sc);
        a[f][8]  = (_Float16)(c1.x * sc);  a[f][9]  = (_Float16)(c1.y * sc);
        a[f][10] = (_Float16)(c1.z * sc);  a[f][11] = (_Float16)(c1.w * sc);
        a[f][12] = (_Float16)(c1b.x * sc); a[f][13] = (_Float16)(c1b.y * sc);
        a[f][14] = (_Float16)(c1b.z * sc); a[f][15] = (_Float16)(c1b.w * sc);
    }

    float* lZw = lZ + wave * 16 * ZSTRIDE;

    // 4 column tiles of 16 outputs each; A fragments reused across tiles
#pragma unroll
    for (int ct = 0; ct < 4; ++ct) {
        const int ncol = ct * 16 + rloc;
        // B fragment from LDS: lane holds col N=ncol; elem j -> K = Kb+16*hi+j
        v16h b[4];
#pragma unroll
        for (int f = 0; f < 4; ++f)
            b[f] = *(const v16h*)(lW + ncol * 128 + f * 32 + 16 * hi);

        const float bv = bias[ncol];
        v8f c;
#pragma unroll
        for (int r = 0; r < 8; ++r) c[r] = bv;

        c = __builtin_amdgcn_wmma_f32_16x16x32_f16(false, a[0], false, b[0], (short)0, c, false, false);
        c = __builtin_amdgcn_wmma_f32_16x16x32_f16(false, a[1], false, b[1], (short)0, c, false, false);
        c = __builtin_amdgcn_wmma_f32_16x16x32_f16(false, a[2], false, b[2], (short)0, c, false, false);
        c = __builtin_amdgcn_wmma_f32_16x16x32_f16(false, a[3], false, b[3], (short)0, c, false, false);

        // C/D layout: VGPR r, lane -> M = r + 8*hi, N = ncol; park in LDS
#pragma unroll
        for (int r = 0; r < 8; ++r)
            lZw[(r + 8 * hi) * ZSTRIDE + ncol] = c[r];
    }

    // Branchless coalesced write-back: 2 lanes per row, 8 x b128 per lane.
    // Same-wave LDS stores/loads are in order (DScnt); rows padded to NPAD.
    {
        const int lrow  = lane >> 1;        // 0..15
        const int lhalf = lane & 1;         // 32-float half of the row
        const float* srow = lZw + lrow * ZSTRIDE + lhalf * 32;
        float*       drow = z + (size_t)(row0 + lrow) * HID + lhalf * 32;
#pragma unroll
        for (int i = 0; i < 8; ++i)
            *(float4*)(drow + i * 4) = *(const float4*)(srow + i * 4);
    }
}

// ---------------------------------------------------------------------------
// BatchNorm stats: per-channel sum & sumsq via LDS reduce + global atomics
// ---------------------------------------------------------------------------
__global__ void bn_stats_kernel(const float* __restrict__ z, float* __restrict__ stats, int n) {
    int c = threadIdx.x & 63;
    int g = threadIdx.x >> 6;          // 4 row groups
    float s = 0.f, q = 0.f;
    for (int r = blockIdx.x * 4 + g; r < n; r += gridDim.x * 4) {
        float v = z[(size_t)r * HID + c];
        s += v; q += v * v;
    }
    __shared__ float ls[256], lq[256];
    ls[threadIdx.x] = s; lq[threadIdx.x] = q;
    __syncthreads();
    if (g == 0) {
        s = ls[c] + ls[c + 64] + ls[c + 128] + ls[c + 192];
        q = lq[c] + lq[c + 64] + lq[c + 128] + lq[c + 192];
        atomicAdd(&stats[c], s);
        atomicAdd(&stats[HID + c], q);
    }
}

__global__ void bn_relu_kernel(const float* __restrict__ z, const float* __restrict__ stats,
                               const float* __restrict__ gam, const float* __restrict__ bet,
                               float* __restrict__ h, int n) {
    int t = blockIdx.x * blockDim.x + threadIdx.x;
    if (t >= n * HID) return;
    int c = t & 63;
    float inv_n = 1.0f / (float)n;
    float m = stats[c] * inv_n;
    float v = stats[HID + c] * inv_n - m * m;
    float rs = rsqrtf(v + BN_EPS);
    float val = (z[t] - m) * rs * gam[c] + bet[c];
    h[t] = val > 0.f ? val : 0.f;
}

// ---------------------------------------------------------------------------
// Global mean pool accumulation
// ---------------------------------------------------------------------------
__global__ void pool_kernel(const float* __restrict__ h, const int* __restrict__ batch,
                            float* __restrict__ gsum, float* __restrict__ gcnt, int n) {
    int t = blockIdx.x * blockDim.x + threadIdx.x;
    if (t >= n * HID) return;
    int nid = t >> 6, c = t & 63;
    int g = batch[nid];
    atomicAdd(&gsum[(size_t)g * HID + c], h[t]);
    if (c == 0) atomicAdd(&gcnt[g], 1.0f);
}

// ---------------------------------------------------------------------------
// MLP head: pooled -> relu(fc1) -> fc2 ; one block per graph
// ---------------------------------------------------------------------------
__global__ void head_kernel(const float* __restrict__ gsum, const float* __restrict__ gcnt,
                            const float* __restrict__ fc1w, const float* __restrict__ fc1b,
                            const float* __restrict__ fc2w, const float* __restrict__ fc2b,
                            float* __restrict__ out) {
    int g = blockIdx.x;
    int o = threadIdx.x;                // 64 threads
    __shared__ float pooled[HID], hid[HID];
    float cnt = fmaxf(gcnt[g], 1.0f);
    pooled[o] = gsum[(size_t)g * HID + o] / cnt;
    __syncthreads();
    float acc = fc1b[o];
#pragma unroll 8
    for (int k = 0; k < HID; ++k) acc += pooled[k] * fc1w[o * HID + k];
    hid[o] = fmaxf(acc, 0.f);
    __syncthreads();
    if (o < 2) {
        float a2 = fc2b[o];
#pragma unroll 8
        for (int k = 0; k < HID; ++k) a2 += hid[k] * fc2w[o * HID + k];
        out[g * 2 + o] = a2;
    }
}

// ---------------------------------------------------------------------------
// Launch
// ---------------------------------------------------------------------------
extern "C" void kernel_launch(void* const* d_in, const int* in_sizes, int n_in,
                              void* d_out, int out_size, void* d_ws, size_t ws_size,
                              hipStream_t stream) {
    const float* x       = (const float*)d_in[0];
    const int*   eidx    = (const int*)  d_in[1];
    const int*   batch   = (const int*)  d_in[2];
    const float* w1l = (const float*)d_in[3];  const float* b1 = (const float*)d_in[4];
    const float* w1r = (const float*)d_in[5];
    const float* bn1g = (const float*)d_in[6]; const float* bn1b = (const float*)d_in[7];
    const float* w2l = (const float*)d_in[8];  const float* b2 = (const float*)d_in[9];
    const float* w2r = (const float*)d_in[10];
    const float* bn2g = (const float*)d_in[11]; const float* bn2b = (const float*)d_in[12];
    const float* w3l = (const float*)d_in[13]; const float* b3 = (const float*)d_in[14];
    const float* w3r = (const float*)d_in[15];
    const float* bn3g = (const float*)d_in[16]; const float* bn3b = (const float*)d_in[17];
    const float* fc1w = (const float*)d_in[18]; const float* fc1b = (const float*)d_in[19];
    const float* fc2w = (const float*)d_in[20]; const float* fc2b = (const float*)d_in[21];
    float* out = (float*)d_out;

    const int* src = eidx;
    const int* dst = eidx + N_EDGES;

    // Workspace carve-up (256B aligned)
    char* base = (char*)d_ws;
    size_t off = 0;
    auto take = [&](size_t bytes) -> char* {
        char* p = base + off;
        off = (off + bytes + 255) & ~(size_t)255;
        return p;
    };
    float*     deg    = (float*)take((size_t)N_NODES * 4);
    float*     msum   = (float*)take((size_t)N_NODES * HID * 4);
    float*     hbuf   = (float*)take((size_t)NPAD * HID * 4);   // padded rows
    float*     zbuf   = (float*)take((size_t)NPAD * HID * 4);   // padded rows
    float*     stats  = (float*)take(2 * HID * 4);
    float*     gsum   = (float*)take((size_t)N_GRAPHS * HID * 4);
    float*     gcnt   = (float*)take(N_GRAPHS * 4);
    _Float16*  Wc2    = (_Float16*)take((size_t)HID * 128 * 2);
    _Float16*  Wc3    = (_Float16*)take((size_t)HID * 128 * 2);

    const int NE_BLK   = (N_EDGES + 255) / 256;
    const int NODE_BLK = (N_NODES + 255) / 256;
    const int NH_BLK   = (N_NODES * HID + 255) / 256;
    const int SC64_BLK = (N_EDGES * 16 + 255) / 256;
    const int GEMM_BLK = NPAD / 128;

    // Pack f16 combined weights for layers 2 & 3
    pack_w_kernel<<<(HID * 128 + 255) / 256, 256, 0, stream>>>(w2l, w2r, Wc2);
    pack_w_kernel<<<(HID * 128 + 255) / 256, 256, 0, stream>>>(w3l, w3r, Wc3);

    // Degrees
    hipMemsetAsync(deg, 0, (size_t)N_NODES * 4, stream);
    degree_count_kernel<<<NE_BLK, 256, 0, stream>>>(dst, deg, N_EDGES);
    deg_inv_kernel<<<NODE_BLK, 256, 0, stream>>>(deg, N_NODES);

    // ---- Layer 1 (IN_DIM=4, scalar) ----
    hipMemsetAsync(msum, 0, (size_t)N_NODES * IN_DIM * 4, stream);
    scatter4_kernel<<<NE_BLK, 256, 0, stream>>>(x, src, dst, msum, N_EDGES);
    sage1_kernel<<<NH_BLK, 256, 0, stream>>>(x, msum, deg, w1l, b1, w1r, zbuf, N_NODES);
    hipMemsetAsync(stats, 0, 2 * HID * 4, stream);
    bn_stats_kernel<<<512, 256, 0, stream>>>(zbuf, stats, N_NODES);
    bn_relu_kernel<<<NH_BLK, 256, 0, stream>>>(zbuf, stats, bn1g, bn1b, hbuf, N_NODES);

    // ---- Layer 2 (WMMA) ----
    hipMemsetAsync(msum, 0, (size_t)N_NODES * HID * 4, stream);
    scatter64_kernel<<<SC64_BLK, 256, 0, stream>>>(hbuf, src, dst, msum, N_EDGES);
    sage_gemm_wmma_kernel<<<GEMM_BLK, 256, 0, stream>>>(msum, deg, hbuf, Wc2, b2, zbuf, N_NODES);
    hipMemsetAsync(stats, 0, 2 * HID * 4, stream);
    bn_stats_kernel<<<512, 256, 0, stream>>>(zbuf, stats, N_NODES);
    bn_relu_kernel<<<NH_BLK, 256, 0, stream>>>(zbuf, stats, bn2g, bn2b, hbuf, N_NODES);

    // ---- Layer 3 (WMMA) ----
    hipMemsetAsync(msum, 0, (size_t)N_NODES * HID * 4, stream);
    scatter64_kernel<<<SC64_BLK, 256, 0, stream>>>(hbuf, src, dst, msum, N_EDGES);
    sage_gemm_wmma_kernel<<<GEMM_BLK, 256, 0, stream>>>(msum, deg, hbuf, Wc3, b3, zbuf, N_NODES);
    hipMemsetAsync(stats, 0, 2 * HID * 4, stream);
    bn_stats_kernel<<<512, 256, 0, stream>>>(zbuf, stats, N_NODES);
    bn_relu_kernel<<<NH_BLK, 256, 0, stream>>>(zbuf, stats, bn3g, bn3b, hbuf, N_NODES);

    // ---- Pool + head ----
    hipMemsetAsync(gsum, 0, (size_t)N_GRAPHS * HID * 4, stream);
    hipMemsetAsync(gcnt, 0, N_GRAPHS * 4, stream);
    pool_kernel<<<NH_BLK, 256, 0, stream>>>(hbuf, batch, gsum, gcnt, N_NODES);
    head_kernel<<<N_GRAPHS, HID, 0, stream>>>(gsum, gcnt, fc1w, fc1b, fc2w, fc2b, out);
}
